// GAT_28157805593321
// MI455X (gfx1250) — matrix operand order repaired and checked by reading, not compiled, not run on hardware
//
#include <hip/hip_runtime.h>
#include <hip/hip_bf16.h>

// Fused GAT pipeline for MI455X (gfx1250, wave32, WMMA).
// One workgroup = 16 samples, 8 waves, all intermediates live in dynamic LDS.
// All matrix work lowered to v_wmma_f32_16x16x32_f16 (f16 in, f32 accum).
// Attention exp() is computed once into an LDS tile in A-fragment layout;
// softmax 1/den is folded into the B (feature) fragments. Conv outputs are
// stored transposed so all non-gather fragment traffic is aligned b128.

typedef __attribute__((ext_vector_type(16))) _Float16 v16h;
typedef __attribute__((ext_vector_type(8)))  _Float16 v8h;
typedef __attribute__((ext_vector_type(8)))  float    v8f;

#define WMMA16(a, b, c) \
  __builtin_amdgcn_wmma_f32_16x16x32_f16(false, (a), false, (b), (short)0, (c), false, false)

__device__ __forceinline__ float leaky(float v) { return v >= 0.f ? v : 0.2f * v; }

__device__ __forceinline__ v16h cat8(v8h lo, v8h hh) {
  return __builtin_shufflevector(lo, hh, 0, 1, 2, 3, 4, 5, 6, 7,
                                 8, 9, 10, 11, 12, 13, 14, 15);
}

// A-fragment (16x32 f16, M=row): lane<16 holds K {k0..k0+15},
// lane>=16 holds K {k0+16..k0+31}; two aligned b128 LDS loads.
__device__ __forceinline__ v16h frag_a_lds(const _Float16* W, int stride, int row,
                                           int k0, int hi) {
  const _Float16* base = W + row * stride + k0 + 16 * hi;
  return cat8(*(const v8h*)base, *(const v8h*)(base + 8));
}

// B-fragment from K-contiguous storage: lane holds K {16*hi .. 16*hi+15} for
// its column; two aligned b128 LDS loads.
__device__ __forceinline__ v16h frag_b_kcontig(const _Float16* rowbase, int hi) {
  const _Float16* p = rowbase + 16 * hi;
  return cat8(*(const v8h*)p, *(const v8h*)(p + 8));
}

__global__ __launch_bounds__(256)
void gat_fused_kernel(const float* __restrict__ x,
                      const float* __restrict__ w1, const float* __restrict__ b1,
                      const float* __restrict__ a11, const float* __restrict__ a21,
                      const float* __restrict__ g1, const float* __restrict__ be1,
                      const float* __restrict__ m1, const float* __restrict__ vv1,
                      const float* __restrict__ w2, const float* __restrict__ b2,
                      const float* __restrict__ a12, const float* __restrict__ a22,
                      const float* __restrict__ g2, const float* __restrict__ be2,
                      const float* __restrict__ m2, const float* __restrict__ vv2,
                      const float* __restrict__ lw, const float* __restrict__ lb,
                      float* __restrict__ out) {
  extern __shared__ char smem[];
  _Float16* w1s = (_Float16*)smem;        // 32 x 288 (K padded 275->288)
  _Float16* w2s = w1s + 9216;             // 32 x 160
  _Float16* XE  = w2s + 5120;             // union: padded input Xs (14400) / Ea[16][32][32]
  _Float16* Cs  = XE + 16384;             // c1T[16][25][32] -> later c2T[2][16][12][32]
  _Float16* h1p = Cs + 13312;             // h1 padded [2][16][32][16]; later Zs[16][384]
  float* s1s  = (float*)(h1p + 16384);    // [2][512]
  float* s2s  = s1s + 1024;               // [2][512]
  float* rden = s2s + 1024;               // [512] reciprocal softmax denom
  float* a1v  = rden + 512; float* a2v  = a1v + 32;
  float* a1v2 = a2v + 32;   float* a2v2 = a1v2 + 32;
  float* sc1  = a2v2 + 32;  float* sh1  = sc1 + 32;
  float* sc2  = sh1 + 32;   float* sh2  = sc2 + 32;
  float* b1v  = sh2 + 32;   float* b2v  = b1v + 32;
  float* lbv  = b2v + 32;                 // 16
  short* ktab1 = (short*)(lbv + 16);      // 288: conv1 k -> patch BYTE offset
  short* ktab2 = ktab1 + 288;             // 160: conv2 k -> patch BYTE offset

  const int tid = threadIdx.x;
  const int wave = tid >> 5;
  const int lane = tid & 31;
  const int lane15 = lane & 15;
  const int hi = lane >> 4;
  const int b0 = blockIdx.x * 16;

  // Warm L2/WGP$ for this block's input tile (emits global_prefetch_b8).
  __builtin_prefetch(x + (size_t)b0 * 1250 + (size_t)tid * 80, 0, 1);

  // ---- Stage 0: params + gather tables -> LDS ----
  for (int i = tid; i < 32 * 288; i += 256) {
    int c = i / 288, k = i - c * 288;
    w1s[i] = (_Float16)((k < 275) ? w1[c * 275 + k] : 0.f);
  }
  for (int i = tid; i < 32 * 160; i += 256) w2s[i] = (_Float16)w2[i];
  for (int k = tid; k < 288; k += 256) {
    int kk = (k < 275) ? k : 0;
    int ic = kk / 55, rem = kk - ic * 55;
    int kh = rem / 11, kw = rem - kh * 11;
    ktab1[k] = (short)((ic * 180 + kh * 36 + kw) * 2);
  }
  if (tid < 160) { int ic = tid / 5, kw = tid - ic * 5; ktab2[tid] = (short)((ic * 16 + kw) * 2); }
  if (tid < 32) {
    a1v[tid]  = (tid < 25) ? a11[tid] : 0.f;
    a2v[tid]  = (tid < 25) ? a21[tid] : 0.f;
    a1v2[tid] = (tid < 12) ? a12[tid] : 0.f;
    a2v2[tid] = (tid < 12) ? a22[tid] : 0.f;
    float s = g1[tid] * rsqrtf(vv1[tid] + 1e-5f);
    sc1[tid] = s; sh1[tid] = be1[tid] - m1[tid] * s;
    s = g2[tid] * rsqrtf(vv2[tid] + 1e-5f);
    sc2[tid] = s; sh2[tid] = be2[tid] - m2[tid] * s;
    b1v[tid] = b1[tid]; b2v[tid] = b2[tid];
    if (tid < 13) lbv[tid] = lb[tid];
  }
  __syncthreads();

  // ---- Layer 1: per half (h=0: cols 0..24, h=1: cols 25..49) ----
  for (int h = 0; h < 2; ++h) {
    _Float16* Xs = XE;
    // circularly padded input tile (width 25 -> 35, stride 36), f32 -> f16.
    for (int row = tid; row < 400; row += 256) {
      int s = row / 25, r = row - s * 25;
      int ic = r / 5, kh = r - ic * 5;
      const float* src = x + (size_t)(b0 + s) * 1250 + ic * 250 + kh * 50 + h * 25;
      _Float16* dst = Xs + s * 900 + ic * 180 + kh * 36;
#pragma unroll
      for (int j = 0; j < 5; ++j)  dst[j] = (_Float16)src[20 + j];   // left wrap
#pragma unroll
      for (int j = 0; j < 25; ++j) dst[5 + j] = (_Float16)src[j];    // center
#pragma unroll
      for (int j = 0; j < 5; ++j)  dst[30 + j] = (_Float16)src[j];   // right wrap
    }
    __syncthreads();

    // conv1 as GEMM: one wave owns a position, B gathered once for both mtiles.
    for (int p = wave; p < 25; p += 8) {
      const char* bbyte = (const char*)(Xs + lane15 * 900 + p);  // sample = lane&15
      v8f acc0 = {}, acc1 = {};
#pragma unroll
      for (int ks = 0; ks < 9; ++ks) {
        const int kb = ks * 32 + 16 * hi;
        v16h bf;
#pragma unroll
        for (int sl = 0; sl < 16; ++sl) {
          int k = kb + sl;
          bf[sl] = (k < 275) ? *(const _Float16*)(bbyte + ktab1[k]) : (_Float16)0.f;
        }
        v16h af0 = frag_a_lds(w1s, 288, lane15,      ks * 32, hi);
        v16h af1 = frag_a_lds(w1s, 288, 16 + lane15, ks * 32, hi);
        acc0 = WMMA16(af0, bf, acc0);
        acc1 = WMMA16(af1, bf, acc1);
      }
      // transposed store: c1T[s][p][c], 8 consecutive channels -> one b128
      _Float16* obase = Cs + lane15 * 800 + p * 32 + 8 * hi;
      v8h pk;
#pragma unroll
      for (int r = 0; r < 8; ++r) pk[r] = (_Float16)(acc0[r] + b1v[8 * hi + r]);
      *(v8h*)obase = pk;
#pragma unroll
      for (int r = 0; r < 8; ++r) pk[r] = (_Float16)(acc1[r] + b1v[16 + 8 * hi + r]);
      *(v8h*)(obase + 16) = pk;
    }
    __syncthreads();

    // s1/s2 node scores from c1T[s][f][j]
    for (int it = tid; it < 512; it += 256) {
      int s = it >> 5, j = it & 31;
      const _Float16* fb = Cs + s * 800 + j;
      float d1 = 0.f, d2 = 0.f;
#pragma unroll
      for (int f = 0; f < 25; ++f) { float v = (float)fb[f * 32]; d1 += v * a1v[f]; d2 += v * a2v[f]; }
      s1s[it] = d1; s2s[it] = d2;
    }
    __syncthreads();

    // E tile: Ea[s][i][j] = exp(leaky(s1[j]+s2[i])), f16, j-contiguous (Xs dead)
    _Float16* Ea = XE;
    for (int it = tid; it < 512; it += 256) {
      int s = it >> 5, i = it & 31;
      float s2i = s2s[s * 32 + i];
      const float* s1p = s1s + s * 32;
      _Float16* er = Ea + s * 1024 + i * 32;
#pragma unroll
      for (int c4 = 0; c4 < 4; ++c4) {
        v8h pk;
#pragma unroll
        for (int q = 0; q < 8; ++q) pk[q] = (_Float16)__expf(leaky(s1p[c4 * 8 + q] + s2i));
        *(v8h*)(er + c4 * 8) = pk;
      }
    }
    __syncthreads();
    // column sums -> reciprocal denominators
    for (int it = tid; it < 512; it += 256) {
      int s = it >> 5, l = it & 31;
      const _Float16* ec = Ea + s * 1024 + l;
      float den = 0.f;
#pragma unroll
      for (int i = 0; i < 32; ++i) den += (float)ec[i * 32];
      rden[it] = 1.f / den;
    }
    __syncthreads();

    // attention einsum (per-sample E(32x32) @ (rden*F)(32x25)) + BN+pool+ReLU
    for (int s = wave * 2; s < wave * 2 + 2; ++s) {
      v8f rv0 = *(const v8f*)(rden + s * 32 + 16 * hi);      // j = 16hi..+7
      v8f rv1 = *(const v8f*)(rden + s * 32 + 16 * hi + 8);  // j = 16hi+8..+15
      v16h bf0 = frag_b_kcontig(Cs + s * 800 + lane15 * 32, hi);
      v16h bf1 = {};
      if (16 + lane15 < 25) bf1 = frag_b_kcontig(Cs + s * 800 + (16 + lane15) * 32, hi);
#pragma unroll
      for (int q = 0; q < 8; ++q) {
        _Float16 r0 = (_Float16)rv0[q], r1 = (_Float16)rv1[q];
        bf0[q] *= r0; bf1[q] *= r0;
        bf0[8 + q] *= r1; bf1[8 + q] *= r1;
      }
#pragma unroll
      for (int mt = 0; mt < 2; ++mt) {
        v16h af = frag_a_lds(Ea + s * 1024, 32, mt * 16 + lane15, 0, hi);
#pragma unroll
        for (int nt = 0; nt < 2; ++nt) {
          int f = nt * 16 + lane15;
          v8f acc = {};
          acc = WMMA16(af, (nt == 0) ? bf0 : bf1, acc);
#pragma unroll
          for (int r = 0; r < 8; ++r) {
            int ii = mt * 16 + r + 8 * hi;
            float y = acc[r] * sc1[ii] + sh1[ii];
            float yn = __shfl_xor(y, 1, 32);  // pair-max pooling across lanes
            if (((f & 1) == 0) && (f < 24)) {
              float pv = fmaxf(fmaxf(y, yn), 0.f);
              int fp = f >> 1;
              _Float16 hv = (_Float16)pv;
              _Float16* hb = h1p + ((h * 16 + s) * 32 + ii) * 16;
              hb[2 + fp] = hv;                 // center (pad offset 2)
              if (fp < 2)  hb[14 + fp] = hv;   // right wrap
              if (fp >= 10) hb[fp - 10] = hv;  // left wrap
            }
          }
        }
      }
    }
    __syncthreads();
  }

  // ---- Layer 2: conv2 on both halves (K=160, 5 steps) ----
  for (int t = wave; t < 24; t += 8) {
    int g = t / 12, p = t - g * 12;
    const char* bbyte = (const char*)(h1p + g * 8192 + lane15 * 512 + p);
    v8f acc0 = {}, acc1 = {};
#pragma unroll
    for (int ks = 0; ks < 5; ++ks) {
      const int kb = ks * 32 + 16 * hi;
      v16h bf;
#pragma unroll
      for (int sl = 0; sl < 16; ++sl) bf[sl] = *(const _Float16*)(bbyte + ktab2[kb + sl]);
      v16h af0 = frag_a_lds(w2s, 160, lane15,      ks * 32, hi);
      v16h af1 = frag_a_lds(w2s, 160, 16 + lane15, ks * 32, hi);
      acc0 = WMMA16(af0, bf, acc0);
      acc1 = WMMA16(af1, bf, acc1);
    }
    // c2T[g][s][p][c]
    _Float16* obase = Cs + ((g * 16 + lane15) * 12 + p) * 32 + 8 * hi;
    v8h pk;
#pragma unroll
    for (int r = 0; r < 8; ++r) pk[r] = (_Float16)(acc0[r] + b2v[8 * hi + r]);
    *(v8h*)obase = pk;
#pragma unroll
    for (int r = 0; r < 8; ++r) pk[r] = (_Float16)(acc1[r] + b2v[16 + 8 * hi + r]);
    *(v8h*)(obase + 16) = pk;
  }
  __syncthreads();

  // layer-2 scores from c2T[g][s][f][j]
  for (int it = tid; it < 1024; it += 256) {
    int g = it >> 9; int rest = it & 511; int s = rest >> 5, j = rest & 31;
    const _Float16* fb = Cs + (g * 16 + s) * 384 + j;
    float d1 = 0.f, d2 = 0.f;
#pragma unroll
    for (int f = 0; f < 12; ++f) { float v = (float)fb[f * 32]; d1 += v * a1v2[f]; d2 += v * a2v2[f]; }
    s1s[it] = d1; s2s[it] = d2;
  }
  __syncthreads();

  // attn2 per combo cb: x2 (cb=0) uses s1 half0 + s2 half1; n2 (cb=1) swapped.
  _Float16* Ea = XE;
  _Float16* Zs = h1p;  // h1 consumed by conv2; z[16][384] lives here now
  for (int cb = 0; cb < 2; ++cb) {
    for (int it = tid; it < 512; it += 256) {
      int s = it >> 5, i = it & 31;
      float s2i = s2s[(cb ^ 1) * 512 + s * 32 + i];
      const float* s1p = s1s + cb * 512 + s * 32;
      _Float16* er = Ea + s * 1024 + i * 32;
#pragma unroll
      for (int c4 = 0; c4 < 4; ++c4) {
        v8h pk;
#pragma unroll
        for (int q = 0; q < 8; ++q) pk[q] = (_Float16)__expf(leaky(s1p[c4 * 8 + q] + s2i));
        *(v8h*)(er + c4 * 8) = pk;
      }
    }
    __syncthreads();
    for (int it = tid; it < 512; it += 256) {
      int s = it >> 5, l = it & 31;
      const _Float16* ec = Ea + s * 1024 + l;
      float den = 0.f;
#pragma unroll
      for (int i = 0; i < 32; ++i) den += (float)ec[i * 32];
      rden[it] = 1.f / den;
    }
    __syncthreads();
    for (int s = wave * 2; s < wave * 2 + 2; ++s) {
      v8f rv0 = *(const v8f*)(rden + s * 32 + 16 * hi);
      v8f rv1 = *(const v8f*)(rden + s * 32 + 16 * hi + 8);
      int f = lane15;
      v16h bf = {};
      if (f < 12) bf = frag_b_kcontig(Cs + ((cb * 16 + s) * 12 + f) * 32, hi);
#pragma unroll
      for (int q = 0; q < 8; ++q) {
        bf[q] *= (_Float16)rv0[q];
        bf[8 + q] *= (_Float16)rv1[q];
      }
#pragma unroll
      for (int mt = 0; mt < 2; ++mt) {
        v16h af = frag_a_lds(Ea + s * 1024, 32, mt * 16 + lane15, 0, hi);
        v8f acc = {};
        acc = WMMA16(af, bf, acc);
#pragma unroll
        for (int r = 0; r < 8; ++r) {
          int ii = mt * 16 + r + 8 * hi;
          float y = acc[r] * sc2[ii] + sh2[ii];
          float yn = __shfl_xor(y, 1, 32);
          if (((f & 1) == 0) && (f < 12)) {
            float pv = fmaxf(fmaxf(y, yn), 0.f);
            Zs[s * 384 + ii * 12 + cb * 6 + (f >> 1)] = (_Float16)pv;
          }
        }
      }
    }
    __syncthreads();
  }

  // ---- Final linear: out[16samp x 13] = z(16x384) @ lw^T, one wave ----
  if (wave == 0) {
    v8f acc = {};
#pragma unroll
    for (int ks = 0; ks < 12; ++ks) {
      int k0 = ks * 32;
      v16h af = frag_a_lds(Zs, 384, lane15, k0, hi);  // row = sample
      v16h bf;
#pragma unroll
      for (int sl = 0; sl < 16; ++sl) {
        int k = k0 + 16 * hi + sl;
        bf[sl] = (lane15 < 13) ? (_Float16)lw[lane15 * 384 + k] : (_Float16)0.f;
      }
      acc = WMMA16(af, bf, acc);
    }
    if (lane15 < 13) {
#pragma unroll
      for (int r = 0; r < 8; ++r) {
        int sm = r + 8 * hi;
        out[(size_t)(b0 + sm) * 13 + lane15] = acc[r] + lbv[lane15];
      }
    }
  }
}

extern "C" void kernel_launch(void* const* d_in, const int* in_sizes, int n_in,
                              void* d_out, int out_size, void* d_ws, size_t ws_size,
                              hipStream_t stream) {
  (void)n_in; (void)d_ws; (void)ws_size; (void)out_size;
  constexpr size_t SMEM =
      (size_t)(9216 + 5120 + 16384 + 13312 + 16384) * 2 +   // f16 regions
      (size_t)(1024 + 1024 + 512 + 10 * 32 + 16) * 4 +       // f32 regions
      (size_t)(288 + 160) * 2;                                // gather tables
  const float* x   = (const float*)d_in[0];
  const float* w1  = (const float*)d_in[1];
  const float* b1  = (const float*)d_in[2];
  const float* a11 = (const float*)d_in[3];
  const float* a21 = (const float*)d_in[4];
  const float* g1  = (const float*)d_in[5];
  const float* be1 = (const float*)d_in[6];
  const float* m1  = (const float*)d_in[7];
  const float* v1  = (const float*)d_in[8];
  const float* w2  = (const float*)d_in[9];
  const float* b2  = (const float*)d_in[10];
  const float* a12 = (const float*)d_in[11];
  const float* a22 = (const float*)d_in[12];
  const float* g2  = (const float*)d_in[13];
  const float* be2 = (const float*)d_in[14];
  const float* m2  = (const float*)d_in[15];
  const float* v2  = (const float*)d_in[16];
  const float* lw  = (const float*)d_in[17];
  const float* lb  = (const float*)d_in[18];
  float* out = (float*)d_out;

  const int B = in_sizes[0] / 1250;   // (B, 5, 5, 50)
  const int grid = B / 16;            // 16 samples per workgroup

  (void)hipFuncSetAttribute(reinterpret_cast<const void*>(&gat_fused_kernel),
                            hipFuncAttributeMaxDynamicSharedMemorySize, (int)SMEM);

  gat_fused_kernel<<<grid, 256, SMEM, stream>>>(
      x, w1, b1, a11, a21, g1, be1, m1, v1,
      w2, b2, a12, a22, g2, be2, m2, v2, lw, lb, out);
}
